// DLRM_1683627180423
// MI455X (gfx1250) — matrix operand order repaired and checked by reading, not compiled, not run on hardware
//
#include <hip/hip_runtime.h>
#include <hip/hip_bf16.h>

typedef _Float16 v16h __attribute__((ext_vector_type(16)));
typedef _Float16 v8h  __attribute__((ext_vector_type(8)));
typedef float    v8f  __attribute__((ext_vector_type(8)));

// ---------------- workspace layout (bytes) ----------------
// f16 weights, [Kp][N] row-major, K zero-padded to multiple of 32
static constexpr size_t OFF_W1 = 0;          //  32*512 *2 =   32768
static constexpr size_t OFF_W2 = 32768;      // 512*256 *2 =  262144
static constexpr size_t OFF_W3 = 294912;     // 256*128 *2 =   65536
static constexpr size_t OFF_W4 = 360448;     // 480*1024*2 =  983040
static constexpr size_t OFF_W5 = 1343488;    //1024*1024*2 = 2097152
static constexpr size_t OFF_W6 = 3440640;    //1024*512 *2 = 1048576
static constexpr size_t OFF_W7 = 4489216;    // 512*256 *2 =  262144
static constexpr size_t OFF_R0 = 8ull  * 1024 * 1024;   // 32MB ping
static constexpr size_t OFF_R1 = 40ull * 1024 * 1024;   // 32MB pong

// ---------------- fragment loaders (ISA 7.12.2 layouts) ----------------
// A 16x32 f16: lane l -> row m=l&15; elems 0-7: K=hi*8+e, elems 8-15: K=16+hi*8+e
__device__ __forceinline__ v16h load_a(const _Float16* p /*row start + kc*32*/, int hi) {
    union { v16h v; v8h h[2]; } u;
    u.h[0] = *(const v8h*)(p + hi * 8);
    u.h[1] = *(const v8h*)(p + 16 + hi * 8);
    return u.v;
}
// B 32x16 f16: lane l -> col n=l&15, elems K as in A; unused helper, reads 16 contiguous n of a W row
__device__ __forceinline__ v16h load_b(const _Float16* p /* &W[k][n_base] */) {
    union { v16h v; v8h h[2]; } u;
    u.h[0] = *(const v8h*)(p);
    u.h[1] = *(const v8h*)(p + 8);
    return u.v;
}
__device__ __forceinline__ v8f wmma16(v16h a, v16h b, v8f c) {
    return __builtin_amdgcn_wmma_f32_16x16x32_f16(false, a, false, b, (short)0, c, false, false);
}

// ---------------- packing kernels ----------------
__global__ void pack_w(const float* __restrict__ W, _Float16* __restrict__ Wp,
                       int K, int N, int Kp) {
    int id = blockIdx.x * 256 + threadIdx.x;
    if (id >= Kp * N) return;
    int k = id / N, n = id - k * N;
    Wp[id] = (k < K) ? (_Float16)W[(size_t)k * N + n] : (_Float16)0.f;
}

__global__ void pack_in(const float* __restrict__ dense, _Float16* __restrict__ A0) {
    int id = blockIdx.x * 256 + threadIdx.x;   // 16384*32 total
    int m = id >> 5, k = id & 31;
    A0[id] = (k < 13) ? (_Float16)dense[m * 13 + k] : (_Float16)0.f;
}

// ---------------- fused GEMM + bias + ReLU ----------------
// C[M][N] = relu(A[M][KP] @ W[KP][N] + bias).
// Wave tile: 32(M) x 64(N): 2 A-frags, 4 B-frags, 8 accumulators.
// Block = 8 waves -> 256(M) x 64(N).  grid = (N/64, M/256).
template<int N, int KP>
__global__ __launch_bounds__(256) void gemm_relu(
    const _Float16* __restrict__ A, const _Float16* __restrict__ W,
    const float* __restrict__ bias, _Float16* __restrict__ C) {
    int lane = threadIdx.x & 31, wave = threadIdx.x >> 5;
    int hi = lane >> 4, lo = lane & 15;
    int m_base = blockIdx.y * 256 + wave * 32;
    int n_base = blockIdx.x * 64;

    const _Float16* arow0 = A + (size_t)(m_base + lo) * KP;
    const _Float16* arow1 = arow0 + (size_t)16 * KP;
    const _Float16* wbase = W + n_base + lo;

    v8f zero = {};
    v8f acc0[4] = {zero, zero, zero, zero};
    v8f acc1[4] = {zero, zero, zero, zero};

#pragma unroll 2
    for (int kc = 0; kc < KP / 32; ++kc) {
        v16h a0 = load_a(arow0 + kc * 32, hi);
        v16h a1 = load_a(arow1 + kc * 32, hi);
        const _Float16* wrow = wbase + (size_t)kc * 32 * N;
        v16h bf[4];
#pragma unroll
        for (int j = 0; j < 4; ++j)
#pragma unroll
            for (int s = 0; s < 16; ++s) bf[j][s] = wrow[(size_t)(hi * 8 + (s & 7) + (s >> 3) * 16) * N + j * 16];
#pragma unroll
        for (int j = 0; j < 4; ++j) {
            acc0[j] = wmma16(a0, bf[j], acc0[j]);
            acc1[j] = wmma16(a1, bf[j], acc1[j]);
        }
    }
#pragma unroll
    for (int j = 0; j < 4; ++j) {
        int n = n_base + j * 16 + lo;
        float bs = bias[n];
#pragma unroll
        for (int r = 0; r < 8; ++r) {
            int m0 = m_base + r + hi * 8;
            float v0 = acc0[j][r] + bs;
            v0 = v0 > 0.f ? v0 : 0.f;
            C[(size_t)m0 * N + n] = (_Float16)v0;
            float v1 = acc1[j][r] + bs;
            v1 = v1 > 0.f ? v1 : 0.f;
            C[(size_t)(m0 + 16) * N + n] = (_Float16)v1;
        }
    }
}

// ---------------- gather + pairwise interactions ----------------
// One wave per sample. combined = [ed_row ; 26 gathered embed rows ; 5 zero rows]
// Z = combined @ combined^T via 3 WMMA tiles; emit x[b] = [ed(128) | triu(351) | pad(1)]
__global__ __launch_bounds__(128) void interact(
    const _Float16* __restrict__ ed, const int* __restrict__ sidx,
    const float* __restrict__ table, _Float16* __restrict__ x) {
    __shared__ _Float16 comb_s[4][32 * 128];
    __shared__ float    zs_s[4][32 * 32];
    int lane = threadIdx.x & 31, wave = threadIdx.x >> 5;
    int hi = lane >> 4, lo = lane & 15;
    int b = blockIdx.x * 4 + wave;
    _Float16* cm = comb_s[wave];
    float*    z  = zs_s[wave];

    // stage combined (32 rows x 128 f16), rows 27..31 zeroed
    for (int r = 0; r < 32; ++r) {
        _Float16* dst = cm + r * 128 + lane * 4;
        if (r == 0) {
            const _Float16* s = ed + (size_t)b * 128 + lane * 4;
            dst[0] = s[0]; dst[1] = s[1]; dst[2] = s[2]; dst[3] = s[3];
        } else if (r <= 26) {
            int f = r - 1;
            long row = (long)sidx[b * 26 + f] + (long)f * 100000;
            const float* s = table + row * 128 + lane * 4;
            dst[0] = (_Float16)s[0]; dst[1] = (_Float16)s[1];
            dst[2] = (_Float16)s[2]; dst[3] = (_Float16)s[3];
        } else {
            dst[0] = dst[1] = dst[2] = dst[3] = (_Float16)0.f;
        }
    }
    __syncthreads();

    v8f t00 = {}, t01 = {}, t11 = {};
    for (int kc = 0; kc < 4; ++kc) {
        v16h a0 = load_a(cm + lo * 128 + kc * 32, hi);
        v16h a1 = load_a(cm + (16 + lo) * 128 + kc * 32, hi);
        v16h b0 = a0, b1 = a1;
        t00 = wmma16(a0, b0, t00);
        t01 = wmma16(a0, b1, t01);
        t11 = wmma16(a1, b1, t11);
    }
#pragma unroll
    for (int r = 0; r < 8; ++r) {
        int m = r + hi * 8, n = lo;
        z[m * 32 + n]             = t00[r];
        z[m * 32 + 16 + n]        = t01[r];
        z[(16 + m) * 32 + 16 + n] = t11[r];
    }
    __syncthreads();

    _Float16* xb = x + (size_t)b * 480;
    for (int kk = lane; kk < 128; kk += 32) xb[kk] = cm[kk];   // ed part
    for (int t = lane; t < 351; t += 32) {                      // row-major triu k=1
        int tt = t, i = 0;
        while (tt >= 26 - i) { tt -= 26 - i; ++i; }
        int j = i + 1 + tt;
        xb[128 + t] = (_Float16)z[i * 32 + j];
    }
    if (lane == 0) xb[479] = (_Float16)0.f;                     // K-pad
}

// ---------------- final 256 -> 1 layer (no ReLU) ----------------
__global__ __launch_bounds__(256) void final_dot(
    const _Float16* __restrict__ act, const float* __restrict__ W,
    const float* __restrict__ bias, float* __restrict__ out) {
    int b = blockIdx.x * 256 + threadIdx.x;
    const _Float16* a = act + (size_t)b * 256;
    float acc = 0.f;
#pragma unroll 8
    for (int k = 0; k < 256; ++k) acc += (float)a[k] * W[k];
    out[b] = acc + bias[0];
}

// ---------------- host ----------------
extern "C" void kernel_launch(void* const* d_in, const int* in_sizes, int n_in,
                              void* d_out, int out_size, void* d_ws, size_t ws_size,
                              hipStream_t stream) {
    const float* dense = (const float*)d_in[0];
    const int*   sidx  = (const int*)d_in[1];
    const float* table = (const float*)d_in[2];
    const float* W1 = (const float*)d_in[3];  const float* b1 = (const float*)d_in[4];
    const float* W2 = (const float*)d_in[5];  const float* b2 = (const float*)d_in[6];
    const float* W3 = (const float*)d_in[7];  const float* b3 = (const float*)d_in[8];
    const float* W4 = (const float*)d_in[9];  const float* b4 = (const float*)d_in[10];
    const float* W5 = (const float*)d_in[11]; const float* b5 = (const float*)d_in[12];
    const float* W6 = (const float*)d_in[13]; const float* b6 = (const float*)d_in[14];
    const float* W7 = (const float*)d_in[15]; const float* b7 = (const float*)d_in[16];
    const float* W8 = (const float*)d_in[17]; const float* b8 = (const float*)d_in[18];

    char* ws = (char*)d_ws;
    _Float16* Wp1 = (_Float16*)(ws + OFF_W1);
    _Float16* Wp2 = (_Float16*)(ws + OFF_W2);
    _Float16* Wp3 = (_Float16*)(ws + OFF_W3);
    _Float16* Wp4 = (_Float16*)(ws + OFF_W4);
    _Float16* Wp5 = (_Float16*)(ws + OFF_W5);
    _Float16* Wp6 = (_Float16*)(ws + OFF_W6);
    _Float16* Wp7 = (_Float16*)(ws + OFF_W7);
    _Float16* R0  = (_Float16*)(ws + OFF_R0);
    _Float16* R1  = (_Float16*)(ws + OFF_R1);

    auto packw = [&](const float* W, _Float16* Wp, int K, int N, int Kp) {
        int tot = Kp * N;
        pack_w<<<(tot + 255) / 256, 256, 0, stream>>>(W, Wp, K, N, Kp);
    };
    packw(W1, Wp1, 13,   512,  32);
    packw(W2, Wp2, 512,  256,  512);
    packw(W3, Wp3, 256,  128,  256);
    packw(W4, Wp4, 479,  1024, 480);
    packw(W5, Wp5, 1024, 1024, 1024);
    packw(W6, Wp6, 1024, 512,  1024);
    packw(W7, Wp7, 512,  256,  512);

    pack_in<<<(16384 * 32) / 256, 256, 0, stream>>>(dense, R0);      // A0 -> R0

    // dense MLP
    gemm_relu<512, 32>  <<<dim3(8, 64),  256, 0, stream>>>(R0, Wp1, b1, R1);  // act1 -> R1
    gemm_relu<256, 512> <<<dim3(4, 64),  256, 0, stream>>>(R1, Wp2, b2, R0);  // act2 -> R0
    gemm_relu<128, 256> <<<dim3(2, 64),  256, 0, stream>>>(R0, Wp3, b3, R1);  // ed   -> R1

    // gather + interactions -> x[16384][480] in R0
    interact<<<16384 / 4, 128, 0, stream>>>(R1, sidx, table, R0);

    // over MLP
    gemm_relu<1024, 480> <<<dim3(16, 64), 256, 0, stream>>>(R0, Wp4, b4, R1); // -> R1
    gemm_relu<1024, 1024><<<dim3(16, 64), 256, 0, stream>>>(R1, Wp5, b5, R0); // -> R0
    gemm_relu<512, 1024> <<<dim3(8, 64),  256, 0, stream>>>(R0, Wp6, b6, R1); // -> R1
    gemm_relu<256, 512>  <<<dim3(4, 64),  256, 0, stream>>>(R1, Wp7, b7, R0); // -> R0

    final_dot<<<16384 / 256, 256, 0, stream>>>(R0, W8, b8, (float*)d_out);
}
